// DecoderLayer_42219528519895
// MI455X (gfx1250) — compile-verified, matrix operand order checked by reading, not run on hardware
//
#include <hip/hip_runtime.h>

// ---------------------------------------------------------------------------
// Types for WMMA (CDNA5 / gfx1250, wave32)
// ---------------------------------------------------------------------------
typedef __attribute__((ext_vector_type(16))) __bf16      v16bf;
typedef __attribute__((ext_vector_type(8)))  float       v8f;
typedef __attribute__((ext_vector_type(4)))  unsigned    u32x4;

union Frag { v16bf v; u32x4 q[2]; };   // 16 bf16 = 8 VGPRs (A/B fragment)

static __device__ __forceinline__ unsigned short f2bf(float f) {
  unsigned u = __float_as_uint(f);
  u += 0x7fffu + ((u >> 16) & 1u);     // round-to-nearest-even
  return (unsigned short)(u >> 16);
}

#define TB   4
#define TS   1024
#define TD   1024
#define TH   16
#define THD  64
#define TDFF 4096
#define TM   (TB * TS)   // 4096 rows

// ---------------------------------------------------------------------------
// fp32 -> bf16 convert (weights / encoder activations)
// ---------------------------------------------------------------------------
__global__ void cvt_bf16_kernel(const float* __restrict__ in,
                                unsigned short* __restrict__ out, int n) {
  int i = blockIdx.x * 256 + threadIdx.x;
  if (i < n) out[i] = f2bf(in[i]);
}

// ---------------------------------------------------------------------------
// LayerNorm (fp32 in) -> bf16 out.  One block (256 thr) per row, D = 1024.
// ---------------------------------------------------------------------------
__global__ __launch_bounds__(256)
void ln_bf16_kernel(const float* __restrict__ x, const float* __restrict__ g,
                    const float* __restrict__ be,
                    unsigned short* __restrict__ out, int D) {
  const int row = blockIdx.x;
  const float* xr = x + (size_t)row * D;
  float s = 0.f, ss = 0.f;
  for (int i = threadIdx.x; i < D; i += 256) {
    float v = xr[i]; s += v; ss += v * v;
  }
  for (int o = 1; o < 32; o <<= 1) {
    s  += __shfl_xor(s,  o, 32);
    ss += __shfl_xor(ss, o, 32);
  }
  __shared__ float rs[8], rss[8];
  if ((threadIdx.x & 31) == 0) { rs[threadIdx.x >> 5] = s; rss[threadIdx.x >> 5] = ss; }
  __syncthreads();
  s = 0.f; ss = 0.f;
  #pragma unroll
  for (int j = 0; j < 8; ++j) { s += rs[j]; ss += rss[j]; }
  const float mu   = s / (float)D;
  const float var  = ss / (float)D - mu * mu;
  const float rstd = rsqrtf(var + 1e-5f);
  for (int i = threadIdx.x; i < D; i += 256) {
    float v = (xr[i] - mu) * rstd * g[i] + be[i];
    out[(size_t)row * D + i] = f2bf(v);
  }
}

// ---------------------------------------------------------------------------
// GEMM: C[M,N] = A[M,K](bf16) @ W[N,K]^T(bf16) + bias (+ReLU)(+resid)
// Block: 256 threads = 8 waves, tile 128x128, K-step 32, 8 wmma / wave / step.
// W row-major (N,K) == B consumed column-major -> same fragment loader as A.
// ---------------------------------------------------------------------------
__global__ __launch_bounds__(256)
void gemm_wmma_kernel(const unsigned short* __restrict__ A,
                      const unsigned short* __restrict__ W,
                      const float* __restrict__ bias,
                      const float* __restrict__ resid,
                      float* __restrict__ outF,
                      unsigned short* __restrict__ outB,
                      int M, int N, int K, int relu) {
  __shared__ __align__(16) unsigned short As[128 * 48];  // stride 48 (96B, 16B-mult)
  __shared__ __align__(16) unsigned short Bs[128 * 48];

  const int tid  = threadIdx.x;
  const int wid  = tid >> 5;
  const int lane = tid & 31;
  const int m16  = lane & 15;
  const int hi   = lane >> 4;
  const int wm   = wid & 3;    // 4 waves over M (32 rows each -> 2 tiles)
  const int wn   = wid >> 2;   // 2 waves over N (64 cols each -> 4 tiles)
  const int rowBase = blockIdx.y * 128;
  const int colBase = blockIdx.x * 128;

  v8f acc[2][4];
  #pragma unroll
  for (int mt = 0; mt < 2; ++mt)
    #pragma unroll
    for (int nt = 0; nt < 4; ++nt)
      #pragma unroll
      for (int e = 0; e < 8; ++e) acc[mt][nt][e] = 0.f;

  const int ldr = tid >> 1;            // 0..127 (tile row this thread stages)
  const int seg = (tid & 1) * 16;      // 16 bf16 = 32B per thread per matrix

  for (int k0 = 0; k0 < K; k0 += 32) {
    const u32x4* ga = (const u32x4*)(A + (size_t)(rowBase + ldr) * K + k0 + seg);
    const u32x4* gb = (const u32x4*)(W + (size_t)(colBase + ldr) * K + k0 + seg);
    u32x4 a0 = ga[0], a1 = ga[1];
    u32x4 b0 = gb[0], b1 = gb[1];
    *(u32x4*)(&As[ldr * 48 + seg])     = a0;
    *(u32x4*)(&As[ldr * 48 + seg + 8]) = a1;
    *(u32x4*)(&Bs[ldr * 48 + seg])     = b0;
    *(u32x4*)(&Bs[ldr * 48 + seg + 8]) = b1;
    __syncthreads();

    Frag af[2], bf[4];
    #pragma unroll
    for (int mt = 0; mt < 2; ++mt) {
      const int r = wm * 32 + mt * 16 + m16;
      af[mt].q[0] = *(const u32x4*)(&As[r * 48 + hi * 8]);
      af[mt].q[1] = *(const u32x4*)(&As[r * 48 + 16 + hi * 8]);
    }
    #pragma unroll
    for (int nt = 0; nt < 4; ++nt) {
      const int r = wn * 64 + nt * 16 + m16;
      bf[nt].q[0] = *(const u32x4*)(&Bs[r * 48 + hi * 8]);
      bf[nt].q[1] = *(const u32x4*)(&Bs[r * 48 + 16 + hi * 8]);
    }
    #pragma unroll
    for (int mt = 0; mt < 2; ++mt)
      #pragma unroll
      for (int nt = 0; nt < 4; ++nt)
        acc[mt][nt] = __builtin_amdgcn_wmma_f32_16x16x32_bf16(
            false, af[mt].v, false, bf[nt].v, (short)0, acc[mt][nt], false, false);
    __syncthreads();
  }

  const bool hasR = (resid != nullptr);
  #pragma unroll
  for (int mt = 0; mt < 2; ++mt)
    #pragma unroll
    for (int nt = 0; nt < 4; ++nt)
      #pragma unroll
      for (int r = 0; r < 8; ++r) {
        const int row = rowBase + wm * 32 + mt * 16 + r + 8 * hi;
        const int col = colBase + wn * 64 + nt * 16 + m16;
        float v = acc[mt][nt][r] + bias[col];
        if (relu) v = fmaxf(v, 0.f);
        if (hasR) v += resid[(size_t)row * N + col];
        if (outF) outF[(size_t)row * N + col] = v;
        if (outB) outB[(size_t)row * N + col] = f2bf(v);
      }
}

// ---------------------------------------------------------------------------
// Flash attention (bf16 QKV, f32 accum).  Layout: [B, S, H*HD] (stride D).
// Block = 128 thr = 4 waves; each wave owns a 16-row query strip, streams
// 32-key blocks.  HD = 64 -> 4 output accumulator tiles per wave.
// ---------------------------------------------------------------------------
__global__ __launch_bounds__(128)
void flash_attn_kernel(const unsigned short* __restrict__ Q,
                       const unsigned short* __restrict__ K,
                       const unsigned short* __restrict__ V,
                       unsigned short* __restrict__ O,
                       int S, int D, int H, int causal) {
  __shared__ __align__(16) unsigned short Pbuf[4][16 * 32];  // 1KB / wave
  __shared__ __align__(16) unsigned short VT[4][64 * 32];    // 4KB / wave (V^T)

  const int wid  = threadIdx.x >> 5;
  const int lane = threadIdx.x & 31;
  const int m    = lane & 15;
  const int hi   = lane >> 4;
  const int bh   = blockIdx.y;
  const int b    = bh / H, h = bh % H;
  const int qrow0 = blockIdx.x * 64 + wid * 16;
  const size_t base = (size_t)b * S * D + (size_t)h * THD;

  // Q fragments (16 rows x 64 HD = 2 A-fragments), resident for whole kernel
  Frag qf[2];
  {
    const unsigned short* qr = Q + base + (size_t)(qrow0 + m) * D;
    #pragma unroll
    for (int j = 0; j < 2; ++j) {
      qf[j].q[0] = *(const u32x4*)(qr + j * 32 + hi * 8);
      qf[j].q[1] = *(const u32x4*)(qr + j * 32 + 16 + hi * 8);
    }
  }

  v8f accO[4];
  float mrow[8], lrow[8];
  #pragma unroll
  for (int v = 0; v < 4; ++v)
    #pragma unroll
    for (int e = 0; e < 8; ++e) accO[v][e] = 0.f;
  #pragma unroll
  for (int r = 0; r < 8; ++r) { mrow[r] = -__builtin_inff(); lrow[r] = 0.f; }

  int nsteps = S >> 5;
  if (causal) nsteps = (qrow0 + 16 + 31) >> 5;

  for (int kb = 0; kb < nsteps; ++kb) {
    const int k0 = kb << 5;

    // stage V^T [HD][32] into this wave's LDS slice (lane owns one key row)
    {
      const unsigned short* vr = V + base + (size_t)(k0 + lane) * D;
      #pragma unroll
      for (int d = 0; d < THD; d += 2) {
        unsigned u = *(const unsigned*)(vr + d);
        VT[wid][d * 32 + lane]       = (unsigned short)(u & 0xffffu);
        VT[wid][(d + 1) * 32 + lane] = (unsigned short)(u >> 16);
      }
    }

    // scores: 2 tiles of 16 keys, each = 2 chained wmma over HD
    v8f s[2];
    #pragma unroll
    for (int t = 0; t < 2; ++t) {
      const unsigned short* kr = K + base + (size_t)(k0 + t * 16 + m) * D;
      Frag k0f, k1f;
      k0f.q[0] = *(const u32x4*)(kr + hi * 8);
      k0f.q[1] = *(const u32x4*)(kr + 16 + hi * 8);
      k1f.q[0] = *(const u32x4*)(kr + 32 + hi * 8);
      k1f.q[1] = *(const u32x4*)(kr + 48 + hi * 8);
      v8f c;
      #pragma unroll
      for (int e = 0; e < 8; ++e) c[e] = 0.f;
      c = __builtin_amdgcn_wmma_f32_16x16x32_bf16(false, qf[0].v, false, k0f.v,
                                                  (short)0, c, false, false);
      c = __builtin_amdgcn_wmma_f32_16x16x32_bf16(false, qf[1].v, false, k1f.v,
                                                  (short)0, c, false, false);
      s[t] = c;
    }

    // online softmax (C layout: row = r + 8*hi, col = lane&15)
    #pragma unroll
    for (int r = 0; r < 8; ++r) {
      const int qg = qrow0 + r + 8 * hi;
      float s0 = s[0][r] * 0.125f;   // 1/sqrt(64)
      float s1 = s[1][r] * 0.125f;
      if (causal) {
        if (k0 + m > qg)      s0 = -__builtin_inff();
        if (k0 + 16 + m > qg) s1 = -__builtin_inff();
      }
      float rm = fmaxf(s0, s1);
      rm = fmaxf(rm, __shfl_xor(rm, 1, 32));
      rm = fmaxf(rm, __shfl_xor(rm, 2, 32));
      rm = fmaxf(rm, __shfl_xor(rm, 4, 32));
      rm = fmaxf(rm, __shfl_xor(rm, 8, 32));
      const float mnew = fmaxf(mrow[r], rm);
      const float corr = __expf(mrow[r] - mnew);
      mrow[r] = mnew;
      const float p0 = __expf(s0 - mnew);
      const float p1 = __expf(s1 - mnew);
      float rsum = p0 + p1;
      rsum += __shfl_xor(rsum, 1, 32);
      rsum += __shfl_xor(rsum, 2, 32);
      rsum += __shfl_xor(rsum, 4, 32);
      rsum += __shfl_xor(rsum, 8, 32);
      lrow[r] = lrow[r] * corr + rsum;
      #pragma unroll
      for (int v = 0; v < 4; ++v) accO[v][r] *= corr;
      Pbuf[wid][(r + 8 * hi) * 32 + m]      = f2bf(p0);
      Pbuf[wid][(r + 8 * hi) * 32 + 16 + m] = f2bf(p1);
    }

    // P (C layout) -> A layout via LDS round trip; then P @ V
    Frag pf;
    pf.q[0] = *(const u32x4*)(&Pbuf[wid][m * 32 + hi * 8]);
    pf.q[1] = *(const u32x4*)(&Pbuf[wid][m * 32 + 16 + hi * 8]);
    #pragma unroll
    for (int v = 0; v < 4; ++v) {
      Frag bf;
      const unsigned short* vt = &VT[wid][(v * 16 + m) * 32];
      bf.q[0] = *(const u32x4*)(vt + hi * 8);
      bf.q[1] = *(const u32x4*)(vt + 16 + hi * 8);
      accO[v] = __builtin_amdgcn_wmma_f32_16x16x32_bf16(false, pf.v, false, bf.v,
                                                        (short)0, accO[v], false, false);
    }
  }

  #pragma unroll
  for (int v = 0; v < 4; ++v)
    #pragma unroll
    for (int r = 0; r < 8; ++r) {
      const int qg = qrow0 + r + 8 * hi;
      float val = accO[v][r] / lrow[r];
      O[base + (size_t)qg * D + v * 16 + m] = f2bf(val);
    }
}

// ---------------------------------------------------------------------------
// Host-side orchestration
// ---------------------------------------------------------------------------
extern "C" void kernel_launch(void* const* d_in, const int* in_sizes, int n_in,
                              void* d_out, int out_size, void* d_ws, size_t ws_size,
                              hipStream_t stream) {
  const float* x    = (const float*)d_in[0];
  const float* enc  = (const float*)d_in[1];
  // d_in[2] encoder_mask (all ones), d_in[3] decoder_mask (tril): handled analytically
  const float* Wk1 = (const float*)d_in[4];  const float* bk1 = (const float*)d_in[5];
  const float* Wp1 = (const float*)d_in[6];  const float* bp1 = (const float*)d_in[7];
  const float* Wk2 = (const float*)d_in[8];  const float* bk2 = (const float*)d_in[9];
  const float* Wp2 = (const float*)d_in[10]; const float* bp2 = (const float*)d_in[11];
  const float* Wf1 = (const float*)d_in[12]; const float* bf1 = (const float*)d_in[13];
  const float* Wf2 = (const float*)d_in[14]; const float* bf2 = (const float*)d_in[15];
  const float* g1  = (const float*)d_in[16]; const float* be1 = (const float*)d_in[17];
  const float* g2  = (const float*)d_in[18]; const float* be2 = (const float*)d_in[19];
  const float* g3  = (const float*)d_in[20]; const float* be3 = (const float*)d_in[21];

  unsigned char* w = (unsigned char*)d_ws;
  auto carve = [&](size_t bytes) -> void* {
    void* p = (void*)w;
    w += (bytes + 255) & ~(size_t)255;
    return p;
  };
  const size_t nDD  = (size_t)TD * TD;      // 1M
  const size_t nDF  = (size_t)TDFF * TD;    // 4M
  const size_t nAct = (size_t)TM * TD;      // 4M
  const size_t nFF  = (size_t)TM * TDFF;    // 16M

  unsigned short* Wk1b = (unsigned short*)carve(nDD * 2);
  unsigned short* Wp1b = (unsigned short*)carve(nDD * 2);
  unsigned short* Wk2b = (unsigned short*)carve(nDD * 2);
  unsigned short* Wp2b = (unsigned short*)carve(nDD * 2);
  unsigned short* Wf1b = (unsigned short*)carve(nDF * 2);
  unsigned short* Wf2b = (unsigned short*)carve(nDF * 2);
  unsigned short* encb = (unsigned short*)carve(nAct * 2);
  unsigned short* hb   = (unsigned short*)carve(nAct * 2);
  unsigned short* P1b  = (unsigned short*)carve(nAct * 2);
  unsigned short* O1b  = (unsigned short*)carve(nAct * 2);  // reused for O2
  float*          x2   = (float*)carve(nAct * 4);
  unsigned short* K2b  = (unsigned short*)carve(nAct * 2);
  unsigned short* QV2b = (unsigned short*)carve(nAct * 2);
  float*          x3   = (float*)carve(nAct * 4);
  unsigned short* F1b  = (unsigned short*)carve(nFF * 2);
  float* out = (float*)d_out;

  auto cvt = [&](const float* src, unsigned short* dst, size_t n) {
    cvt_bf16_kernel<<<dim3((unsigned)((n + 255) / 256)), dim3(256), 0, stream>>>(src, dst, (int)n);
  };
  cvt(Wk1, Wk1b, nDD); cvt(Wp1, Wp1b, nDD);
  cvt(Wk2, Wk2b, nDD); cvt(Wp2, Wp2b, nDD);
  cvt(Wf1, Wf1b, nDF); cvt(Wf2, Wf2b, nDF);
  cvt(enc, encb, nAct);

  auto gemm = [&](const unsigned short* A, const unsigned short* W_, const float* bias,
                  const float* resid, float* oF, unsigned short* oB,
                  int M, int N, int K, int relu) {
    gemm_wmma_kernel<<<dim3(N / 128, M / 128), dim3(256), 0, stream>>>(
        A, W_, bias, resid, oF, oB, M, N, K, relu);
  };
  auto flash = [&](const unsigned short* Qp, const unsigned short* Kp,
                   const unsigned short* Vp, unsigned short* Op, int causal) {
    flash_attn_kernel<<<dim3(TS / 64, TB * TH), dim3(128), 0, stream>>>(
        Qp, Kp, Vp, Op, TS, TD, TH, causal);
  };

  // --- masked self-attention (q=k=v=proj(LN(x)) via Wk1, per source bug) ---
  ln_bf16_kernel<<<dim3(TM), dim3(256), 0, stream>>>(x, g1, be1, hb, TD);
  gemm(hb, Wk1b, bk1, nullptr, nullptr, P1b, TM, TD, TD, 0);
  flash(P1b, P1b, P1b, O1b, 1);
  gemm(O1b, Wp1b, bp1, x, x2, nullptr, TM, TD, TD, 0);       // x2 = x + proj

  // --- cross-attention (k=proj(LN(x2)), q=v=proj(enc), all via Wk2) ---
  ln_bf16_kernel<<<dim3(TM), dim3(256), 0, stream>>>(x2, g2, be2, hb, TD);
  gemm(hb, Wk2b, bk2, nullptr, nullptr, K2b, TM, TD, TD, 0);
  gemm(encb, Wk2b, bk2, nullptr, nullptr, QV2b, TM, TD, TD, 0);
  flash(QV2b, K2b, QV2b, O1b, 0);                            // encoder_mask==1
  gemm(O1b, Wp2b, bp2, x2, x3, nullptr, TM, TD, TD, 0);      // x3 = x2 + proj

  // --- feed-forward ---
  ln_bf16_kernel<<<dim3(TM), dim3(256), 0, stream>>>(x3, g3, be3, hb, TD);
  gemm(hb, Wf1b, bf1, nullptr, nullptr, F1b, TM, TDFF, TD, 1);   // ReLU
  gemm(F1b, Wf2b, bf2, x3, out, nullptr, TM, TD, TDFF, 0);       // final f32 out

  (void)in_sizes; (void)n_in; (void)out_size; (void)ws_size;
}
